// VectorQuantizer_62474594287943
// MI455X (gfx1250) — matrix operand order repaired and checked by reading, not compiled, not run on hardware
//
#include <hip/hip_runtime.h>
#include <hip/hip_bf16.h>

typedef _Float16 v8h  __attribute__((ext_vector_type(8)));
typedef _Float16 v16h __attribute__((ext_vector_type(16)));
typedef float    v8f  __attribute__((ext_vector_type(8)));

#define N_EMBED   8192
#define EMBED_DIM 64
#define NTOT      32768          // query vectors (8*64*64)
#define NELEM     2097152        // total elements 8*64*64*64

static __device__ __forceinline__ v16h cat8(v8h lo, v8h hi) {
    return __builtin_shufflevector(lo, hi, 0,1,2,3,4,5,6,7,8,9,10,11,12,13,14,15);
}

// ---------------------------------------------------------------------------
// Kernel 1: z [B,C,H,W] f32  ->  z16 [N, 64] f16 (row n = b*4096 + h*64 + w)
// LDS tile transpose so both global read and write are coalesced.
// ---------------------------------------------------------------------------
__global__ __launch_bounds__(256) void vq_prep_z(const float* __restrict__ z,
                                                 _Float16* __restrict__ z16) {
    __shared__ _Float16 tile[64][65];
    const int bh = blockIdx.x;            // 0..511
    const int b  = bh >> 6;
    const int h  = bh & 63;
    const float* zp = z + (size_t)b * 262144 + (size_t)h * 64;
#pragma unroll
    for (int it = 0; it < 16; ++it) {
        int j = it * 256 + threadIdx.x;   // j over (c,w)
        int c = j >> 6, w = j & 63;
        tile[c][w] = (_Float16)zp[(size_t)c * 4096 + w];
    }
    __syncthreads();
    _Float16* op = z16 + ((size_t)b * 4096 + (size_t)h * 64) * 64;
#pragma unroll
    for (int it = 0; it < 16; ++it) {
        int j = it * 256 + threadIdx.x;   // j over (w,c)
        int w = j >> 6, c = j & 63;
        op[(size_t)w * 64 + c] = tile[c][w];
    }
}

// ---------------------------------------------------------------------------
// Kernel 2: codebook -> f16 scaled by 8192 (keeps values in fp16 normal range)
// plus transformed norm  en2[k] = 1024*|e_k|^2 + 16  so the argmin kernel's
// score  s = en2 - 0.25*dot  is one fmamk and always a positive float
// (u32-orderable). 0.25 == 1024 * (2/8192) exactly.
// ---------------------------------------------------------------------------
__global__ __launch_bounds__(64) void vq_prep_e(const float* __restrict__ emb,
                                                _Float16* __restrict__ e16,
                                                float* __restrict__ enorm2) {
    __shared__ float sd[64];
    const int k = blockIdx.x;
    const int c = threadIdx.x;
    float w = emb[(size_t)k * 64 + c];
    e16[(size_t)k * 64 + c] = (_Float16)(w * 8192.0f);
    sd[c] = w * w;
    __syncthreads();
#pragma unroll
    for (int o = 32; o > 0; o >>= 1) {
        if (c < o) sd[c] += sd[c + o];
        __syncthreads();
    }
    if (c == 0) enorm2[k] = 1024.0f * sd[0] + 16.0f;
}

// ---------------------------------------------------------------------------
// Kernel 3: WMMA distance GEMM + argmin.
// One wave owns 32 query rows (two 16-row M tiles, A register-resident; every
// B fragment feeds 2 WMMAs). B fragments double-buffered: issue loads for
// iteration i+1, run WMMAs for i from registers.
// Argmin at 3 VALU/score: s = fma(dot,-0.25,en2) (positive float) ->
// key = (bits(s) & ~0x1FF) | (kb>>4)  (lane column l is implicit: entries
// seen by a lane are ki = kb + l, so only the 9-bit tile tag is needed) ->
// v_min_u32 running reduce. Ties pick the smaller tag = first occurrence.
// ---------------------------------------------------------------------------
__global__ __launch_bounds__(256) void vq_argmin(const _Float16* __restrict__ z16,
                                                 const _Float16* __restrict__ e16,
                                                 const float* __restrict__ enorm2,
                                                 int* __restrict__ idx) {
    const int lane = threadIdx.x & 31;
    const int wave = threadIdx.x >> 5;
    const int l    = lane & 15;          // A: row M   B/C/D: column N
    const int hi   = lane >> 4;          // K-slice selector
    const int m0   = (blockIdx.x * 8 + wave) * 32;

    // --- A fragments: ISA 16-bit A 16x32 layout.
    const _Float16* za = z16 + (size_t)(m0 + l) * 64;
    const _Float16* zb = za + 16 * 64;   // second 16-row tile
    v16h A0a = cat8(*(const v8h*)(za +      hi * 8), *(const v8h*)(za + 16 + hi * 8));
    v16h A1a = cat8(*(const v8h*)(za + 32 + hi * 8), *(const v8h*)(za + 48 + hi * 8));
    v16h A0b = cat8(*(const v8h*)(zb +      hi * 8), *(const v8h*)(zb + 16 + hi * 8));
    v16h A1b = cat8(*(const v8h*)(zb + 32 + hi * 8), *(const v8h*)(zb + 48 + hi * 8));

    unsigned int besta[8], bestb[8];
#pragma unroll
    for (int r = 0; r < 8; ++r) { besta[r] = 0xFFFFFFFFu; bestb[r] = 0xFFFFFFFFu; }

    // --- B fragment prologue (kb = 0): ISA 16-bit B 32x16 layout.
    const _Float16* eb0 = e16 + (size_t)l * 64 + hi * 8;
    v16h B0 = cat8(*(const v8h*)(eb0),      *(const v8h*)(eb0 + 16));
    v16h B1 = cat8(*(const v8h*)(eb0 + 32), *(const v8h*)(eb0 + 48));
    float en = enorm2[l];

#pragma unroll 2
    for (int kb = 0; kb < N_EMBED; kb += 16) {
        // Issue next tile's loads first (wrap keeps flow uniform; EXEC all-1s).
        const int nkb = (kb + 16) & (N_EMBED - 1);
        const _Float16* eb = e16 + (size_t)(nkb + l) * 64 + hi * 8;
        __builtin_prefetch(eb + 16 * 64, 0, 1);          // global_prefetch_b8
        v8h nb0l = *(const v8h*)(eb);
        v8h nb0h = *(const v8h*)(eb + 16);
        v8h nb1l = *(const v8h*)(eb + 32);
        v8h nb1h = *(const v8h*)(eb + 48);
        float nen = enorm2[nkb + l];

        // Current tile: 4 WMMAs entirely from registers.
        v8f acca = {};
        acca = __builtin_amdgcn_wmma_f32_16x16x32_f16(false, A0a, false, B0,
                                                      (short)0, acca, false, false);
        acca = __builtin_amdgcn_wmma_f32_16x16x32_f16(false, A1a, false, B1,
                                                      (short)0, acca, false, false);
        v8f accb = {};
        accb = __builtin_amdgcn_wmma_f32_16x16x32_f16(false, A0b, false, B0,
                                                      (short)0, accb, false, false);
        accb = __builtin_amdgcn_wmma_f32_16x16x32_f16(false, A1b, false, B1,
                                                      (short)0, accb, false, false);

        const unsigned int tag = (unsigned int)(kb >> 4);   // 9-bit tile id
#pragma unroll
        for (int r = 0; r < 8; ++r) {
            float sa = __builtin_fmaf(acca[r], -0.25f, en);   // always > 0
            unsigned int ka = (__float_as_uint(sa) & 0xFFFFFE00u) | tag;
            besta[r] = min(besta[r], ka);                     // v_min_u32
            float sb = __builtin_fmaf(accb[r], -0.25f, en);
            unsigned int kbk = (__float_as_uint(sb) & 0xFFFFFE00u) | tag;
            bestb[r] = min(bestb[r], kbk);
        }

        // Rotate double buffer.
        B0 = cat8(nb0l, nb0h);
        B1 = cat8(nb1l, nb1h);
        en = nen;
    }

    // Unpack (value, index): entry = tag*16 + l (lane column is implicit).
    unsigned int va[8], vb[8];
    int ia[8], ib[8];
#pragma unroll
    for (int r = 0; r < 8; ++r) {
        va[r] = besta[r] & 0xFFFFFE00u;
        ia[r] = (int)((besta[r] & 511u) << 4) | l;
        vb[r] = bestb[r] & 0xFFFFFE00u;
        ib[r] = (int)((bestb[r] & 511u) << 4) | l;
    }

    // Cross-lane argmin inside each 16-lane half (row M = r + hi*8 lives there).
#pragma unroll
    for (int m = 1; m < 16; m <<= 1) {
#pragma unroll
        for (int r = 0; r < 8; ++r) {
            unsigned int ov = __shfl_xor(va[r], m, 32);
            int          oi = __shfl_xor(ia[r], m, 32);
            if (ov < va[r] || (ov == va[r] && oi < ia[r])) { va[r] = ov; ia[r] = oi; }
            unsigned int ow = __shfl_xor(vb[r], m, 32);
            int          oj = __shfl_xor(ib[r], m, 32);
            if (ow < vb[r] || (ow == vb[r] && oj < ib[r])) { vb[r] = ow; ib[r] = oj; }
        }
    }
    if (l == 0) {
#pragma unroll
        for (int r = 0; r < 8; ++r) {
            idx[m0 +      hi * 8 + r] = ia[r];
            idx[m0 + 16 + hi * 8 + r] = ib[r];
        }
    }
}

// ---------------------------------------------------------------------------
// Kernel 4: gather exact f32 codebook rows (z_q_st == z_q in value) through an
// LDS tile so codebook reads, z reads, and output writes are all coalesced;
// per-block partial sums of (z_q - z)^2.
// ---------------------------------------------------------------------------
__global__ __launch_bounds__(256) void vq_gather(const float* __restrict__ z,
                                                 const float* __restrict__ emb,
                                                 const int* __restrict__ idx,
                                                 float* __restrict__ out,
                                                 float* __restrict__ partials) {
    __shared__ float tile[64][65];
    __shared__ float sd[256];
    const int bh = blockIdx.x;            // 0..511
    const int b  = bh >> 6;
    const int h  = bh & 63;
    const int n0 = (b << 12) + (h << 6);
#pragma unroll
    for (int it = 0; it < 16; ++it) {
        int j = it * 256 + threadIdx.x;   // j over (w,c): c fastest -> row-contig
        int w = j >> 6, c = j & 63;
        int k = idx[n0 + w];              // broadcast within each 64-thread group
        tile[c][w] = emb[(size_t)k * 64 + c];
    }
    __syncthreads();
    float sum = 0.0f;
    const float* zp = z   + (size_t)b * 262144 + (size_t)h * 64;
    float*       op = out + (size_t)b * 262144 + (size_t)h * 64;
#pragma unroll
    for (int it = 0; it < 16; ++it) {
        int j = it * 256 + threadIdx.x;   // j over (c,w): w fastest -> coalesced
        int c = j >> 6, w = j & 63;
        float zq = tile[c][w];
        float zv = zp[(size_t)c * 4096 + w];
        op[(size_t)c * 4096 + w] = zq;
        float d = zq - zv;
        sum += d * d;
    }
    sd[threadIdx.x] = sum;
    __syncthreads();
#pragma unroll
    for (int o = 128; o > 0; o >>= 1) {
        if (threadIdx.x < o) sd[threadIdx.x] += sd[threadIdx.x + o];
        __syncthreads();
    }
    if (threadIdx.x == 0) partials[bh] = sd[0];
}

// ---------------------------------------------------------------------------
// Kernel 5: deterministic fixed-order reduction of 512 partials -> loss.
// loss = mean + beta*mean = 2 * sum / NELEM.
// ---------------------------------------------------------------------------
__global__ __launch_bounds__(256) void vq_loss(const float* __restrict__ partials,
                                               float* __restrict__ loss) {
    __shared__ float sd[256];
    const int t = threadIdx.x;
    float s = partials[t] + partials[t + 256];
    sd[t] = s;
    __syncthreads();
#pragma unroll
    for (int o = 128; o > 0; o >>= 1) {
        if (t < o) sd[t] += sd[t + o];
        __syncthreads();
    }
    if (t == 0) *loss = 2.0f * sd[0] / (float)NELEM;
}

extern "C" void kernel_launch(void* const* d_in, const int* in_sizes, int n_in,
                              void* d_out, int out_size, void* d_ws, size_t ws_size,
                              hipStream_t stream) {
    (void)in_sizes; (void)n_in; (void)out_size; (void)ws_size;
    const float* z   = (const float*)d_in[0];   // [8,64,64,64] f32
    const float* emb = (const float*)d_in[1];   // [8192,64]    f32
    float* out  = (float*)d_out;                // [8,64,64,64] + [loss]
    float* loss = out + NELEM;

    char* ws = (char*)d_ws;
    _Float16* z16    = (_Float16*)(ws);                      // 4,194,304 B
    _Float16* e16    = (_Float16*)(ws + 4194304);            // 1,048,576 B
    float*    enorm2 = (float*)   (ws + 5242880);            //    32,768 B
    int*      idx    = (int*)     (ws + 5275648);            //   131,072 B
    float*    parts  = (float*)   (ws + 5406720);            //     2,048 B

    vq_prep_z <<<512,  256, 0, stream>>>(z, z16);
    vq_prep_e <<<8192,  64, 0, stream>>>(emb, e16, enorm2);
    vq_argmin <<<128,  256, 0, stream>>>(z16, e16, enorm2, idx);  // 1024 waves * 32 rows
    vq_gather <<<512,  256, 0, stream>>>(z, emb, idx, out, parts);
    vq_loss   <<<1,    256, 0, stream>>>(parts, loss);
}
